// SelfAttention_18683107738058
// MI455X (gfx1250) — matrix operand order repaired and checked by reading, not compiled, not run on hardware
//
#include <hip/hip_runtime.h>
#include <hip/hip_bf16.h>

typedef __attribute__((ext_vector_type(16))) _Float16 v16h;
typedef __attribute__((ext_vector_type(8)))  _Float16 v8h;
typedef __attribute__((ext_vector_type(8)))  float    v8f;
typedef __attribute__((ext_vector_type(4)))  float    f4;
typedef __attribute__((ext_vector_type(4)))  int      v4i;

#define BATCH 8
#define CH    256
#define CQK   32
#define NPIX  4096
#define NT    (NPIX / 16)   // 256 query/column tiles per batch

// CDNA5 async global->LDS copy path (ASYNCcnt) if the toolchain exposes it.
#if __has_builtin(__builtin_amdgcn_global_load_async_to_lds_b128) && \
    __has_builtin(__builtin_amdgcn_s_wait_asynccnt)
#define USE_ASYNC_LDS 1
#else
#define USE_ASYNC_LDS 0
#endif

#if __has_builtin(__builtin_amdgcn_exp2f)
#define EXP2F(x) __builtin_amdgcn_exp2f(x)
#else
#define EXP2F(x) exp2f(x)
#endif

// 16-bit A-matrix 16x32 layout (CDNA5 ISA): VGPR v holds K pair
//   v<4 : K = 8*hi + 2v, +1      v>=4 : K = 16 + 8*hi + 2(v-4), +1
__device__ __forceinline__ int apat(int v, int hi) {
    return (v < 4) ? (8 * hi + 2 * v) : (16 + 8 * hi + 2 * (v - 4));
}

__device__ __forceinline__ v8f wmma_f16(v16h a, v16h b, v8f c) {
    return __builtin_amdgcn_wmma_f32_16x16x32_f16(false, a, false, b, (short)0, c,
                                                  false, false);
}

// 16-byte global->LDS copy: async tensor-path copy when available.
// Builtin params are AS1 / AS3 qualified v4i pointers (per hipcc diagnostic).
__device__ __forceinline__ void copy16(const _Float16* g, _Float16* l) {
#if USE_ASYNC_LDS
    __builtin_amdgcn_global_load_async_to_lds_b128(
        (__attribute__((address_space(1))) v4i*)g,
        (__attribute__((address_space(3))) v4i*)l, 0, 0);
#else
    *(v8h*)l = *(const v8h*)g;
#endif
}

// Assemble a v16h from a 16B-aligned (not necessarily 32B) address.
__device__ __forceinline__ v16h load_v16h_16al(const _Float16* p) {
    union { v16h h; v8h q[2]; } u;
    u.q[0] = *(const v8h*)p;
    u.q[1] = *(const v8h*)(p + 8);
    return u.h;
}

// ---------------------------------------------------------------------------
// Stage 1: q/k/v 1x1-conv projections via WMMA f16 (f32 accumulate).
// One wave handles a 16-column (n) tile for ALL 20 output row-tiles
// (q: 2, k: 2, v: 16), reusing the 8 cached x B-tiles (256 x 16 slab).
// Outputs (f16):
//   qT [B][N][32]   (A-operand friendly row-major per query)
//   kF [B][32][N]   (B-operand for S = Q*K^T: lane=d, 16 contiguous keys)
//   vT [B][N][256]  (B-operand for O += P*V: lane=key, 16 contiguous chans)
// ---------------------------------------------------------------------------
__global__ __launch_bounds__(128, 1)
void qkv_proj_kernel(const float* __restrict__ x,
                     const float* __restrict__ Wq, const float* __restrict__ bq,
                     const float* __restrict__ Wk, const float* __restrict__ bk,
                     const float* __restrict__ Wv, const float* __restrict__ bv,
                     _Float16* __restrict__ qT, _Float16* __restrict__ kF,
                     _Float16* __restrict__ vT)
{
    const int lane = threadIdx.x & 31;
    const int lo = lane & 15, hi = lane >> 4;
    const int wid = blockIdx.x * (blockDim.x >> 5) + (threadIdx.x >> 5);
    const int b  = wid >> 8;
    const int nt = wid & 255;
    const int n0 = nt * 16;
    const float* xb = x + (size_t)b * CH * NPIX;

    // Cache the full x column slab [256 x 16] as 8 f16 B-tiles (32x16 each).
    v16h bt[8];
#pragma unroll
    for (int s = 0; s < 8; ++s) {
        v16h t;
#pragma unroll
        for (int e = 0; e < 16; ++e) {
            const int kk = (e < 8) ? (8 * hi + e) : (16 + 8 * hi + (e - 8));
            t[e] = (_Float16)xb[(size_t)(s * 32 + kk) * NPIX + n0 + lo];
        }
        bt[s] = t;
    }

#pragma unroll
    for (int ot = 0; ot < 20; ++ot) {
        const float* Wm; const float* bias; int obase;
        if (ot < 2)      { Wm = Wq; bias = bq; obase = ot * 16; }
        else if (ot < 4) { Wm = Wk; bias = bk; obase = (ot - 2) * 16; }
        else             { Wm = Wv; bias = bv; obase = (ot - 4) * 16; }

        v8f acc = {};
#pragma unroll
        for (int s = 0; s < 8; ++s) {
            const float* wrow = Wm + (size_t)(obase + lo) * CH + s * 32;
            v16h at;
#pragma unroll
            for (int v = 0; v < 8; ++v) {
                const int cp = apat(v, hi);
                at[2 * v]     = (_Float16)wrow[cp];
                at[2 * v + 1] = (_Float16)wrow[cp + 1];
            }
            acc = wmma_f16(at, bt[s], acc);
        }
        // D layout: lane column n = n0+lo ; rows o = obase + 8*hi + v
        if (ot < 2) {
            _Float16* dst = qT + ((size_t)b * NPIX + n0 + lo) * CQK + obase + 8 * hi;
            v8h o16;
#pragma unroll
            for (int v = 0; v < 8; ++v)
                o16[v] = (_Float16)(acc[v] + bias[obase + 8 * hi + v]);
            *(v8h*)dst = o16;
        } else if (ot < 4) {
#pragma unroll
            for (int v = 0; v < 8; ++v) {
                const int o = obase + 8 * hi + v;
                kF[((size_t)b * CQK + o) * NPIX + n0 + lo] =
                    (_Float16)(acc[v] + bias[o]);
            }
        } else {
            _Float16* dst = vT + ((size_t)b * NPIX + n0 + lo) * CH + obase + 8 * hi;
            v8h o16;
#pragma unroll
            for (int v = 0; v < 8; ++v)
                o16[v] = (_Float16)(acc[v] + bias[obase + 8 * hi + v]);
            *(v8h*)dst = o16;
        }
    }
}

// ---------------------------------------------------------------------------
// Stage 2: flash attention. 4 waves/WG, all on the same batch -> K/V chunks
// are shared: staged once per WG into LDS via async global->LDS copies
// (24KB of the 320KB WGP LDS), 4x less L2 traffic than per-wave streaming.
// One wave owns 16 query rows, 16x256 f32 accumulators (128 VGPRs, fine in
// wave32's 1024-VGPR budget). Softmax runs in exp2 domain (log2e folded into
// the Q scale -> raw v_exp_f32), row stats via shfl_xor in 16-lane
// half-groups, and the O-rescale is skipped (wave-uniform __any branch)
// unless the running max actually moved (~10 of 128 chunks).
// ---------------------------------------------------------------------------
__global__ __launch_bounds__(128, 1)
void flash_attn_kernel(const float* __restrict__ x,
                       const _Float16* __restrict__ qT,
                       const _Float16* __restrict__ kF,
                       const _Float16* __restrict__ vT,
                       const float* __restrict__ gamma,
                       float* __restrict__ out)
{
    __shared__ _Float16 lds_k[CQK][40];      // 32(d) x 32(keys) + pad
    __shared__ _Float16 lds_v[32][264];      // 32(keys) x 256(ch) + pad
    __shared__ _Float16 lds_p[4][16][34];    // per-wave P-tile transpose slice

    const int tid  = threadIdx.x;
    const int lane = tid & 31;
    const int lo = lane & 15, hi = lane >> 4;
    const int w = tid >> 5;
    const int wid = blockIdx.x * 4 + w;
    const int b  = wid >> 8;
    const int nt = wid & 255;
    const int n0 = nt * 16;

    // 32^-0.5 * log2(e): scores come out of WMMA already in exp2 domain.
    const float qscale = 0.1767766952966369f * 1.4426950408889634f;

    // Q A-tile (16 rows x 32 d), prescaled.
    v16h qa;
    {
        const _Float16* qrow = qT + ((size_t)b * NPIX + n0 + lo) * CQK;
#pragma unroll
        for (int v = 0; v < 8; ++v) {
            const int cp = apat(v, hi);
            qa[2 * v]     = (_Float16)((float)qrow[cp]     * qscale);
            qa[2 * v + 1] = (_Float16)((float)qrow[cp + 1] * qscale);
        }
    }

    v8f O[16];
#pragma unroll
    for (int t = 0; t < 16; ++t) O[t] = (v8f){};
    float mrow[8], lrow[8];
#pragma unroll
    for (int v = 0; v < 8; ++v) { mrow[v] = -3.0e38f; lrow[v] = 0.0f; }

    const _Float16* kb = kF + (size_t)b * CQK * NPIX;
    const _Float16* vb = vT + (size_t)b * NPIX * CH;

    // Cooperative-copy coordinates (128 threads, 18KB per chunk).
    const int kd = tid >> 2, kpart = tid & 3;      // K: 32 rows x 4 x 16B
    const int vj = tid & 31, vseg = tid >> 5;      // V: 32 rows x 4 segs x 128B

    for (int j0 = 0; j0 < NPIX; j0 += 32) {
        __syncthreads();   // previous chunk's LDS reads complete
        copy16(kb + (size_t)kd * NPIX + j0 + kpart * 8, &lds_k[kd][kpart * 8]);
        {
            const _Float16* vsrc = vb + (size_t)(j0 + vj) * CH + vseg * 64;
            _Float16* vdst = &lds_v[vj][vseg * 64];
#pragma unroll
            for (int i = 0; i < 8; ++i)
                copy16(vsrc + i * 8, vdst + i * 8);
        }
        if (j0 + 32 < NPIX) {
            __builtin_prefetch(vb + (size_t)(j0 + 32 + vj) * CH + vseg * 64, 0, 0);
            __builtin_prefetch(kb + (size_t)kd * NPIX + j0 + 32, 0, 0);
        }
#if USE_ASYNC_LDS
        __builtin_amdgcn_s_wait_asynccnt(0);
#endif
        __syncthreads();

        // S = Q*K^T (scores in exp2 domain); B-tiles from LDS: lane = key, elements = d.
        v16h kb0, kb1;
#pragma unroll
        for (int e = 0; e < 16; ++e) {
            const int kk = (e < 8) ? (8 * hi + e) : (16 + 8 * hi + (e - 8));
            kb0[e] = lds_k[kk][lo];
            kb1[e] = lds_k[kk][16 + lo];
        }
        v8f zero = {};
        v8f s0 = wmma_f16(qa, kb0, zero);
        v8f s1 = wmma_f16(qa, kb1, zero);

        // Per-row chunk max (16-lane half-group all-reduce).
        float pm[8];
#pragma unroll
        for (int v = 0; v < 8; ++v) {
            float t = fmaxf(s0[v], s1[v]);
#pragma unroll
            for (int m = 1; m < 16; m <<= 1)
                t = fmaxf(t, __shfl_xor(t, m, 32));
            pm[v] = t;
        }

        // Rescale only when the running max moved (wave-uniform branch).
        bool chg = false;
#pragma unroll
        for (int v = 0; v < 8; ++v) chg = chg || (pm[v] > mrow[v]);
        if (__any(chg)) {
#pragma unroll
            for (int v = 0; v < 8; ++v) {
                const float mn   = fmaxf(mrow[v], pm[v]);
                const float corr = EXP2F(mrow[v] - mn);
                mrow[v] = mn;
                lrow[v] *= corr;
#pragma unroll
                for (int t = 0; t < 16; ++t) O[t][v] *= corr;
            }
        }

        // P = exp2(S - m); stash f16 tile to per-wave LDS slice; row sums.
        float psum[8];
#pragma unroll
        for (int v = 0; v < 8; ++v) {
            const float e0 = EXP2F(s0[v] - mrow[v]);
            const float e1 = EXP2F(s1[v] - mrow[v]);
            lds_p[w][8 * hi + v][lo]      = (_Float16)e0;
            lds_p[w][8 * hi + v][16 + lo] = (_Float16)e1;
            float ps = e0 + e1;
#pragma unroll
            for (int m = 1; m < 16; m <<= 1)
                ps += __shfl_xor(ps, m, 32);
            psum[v] = ps;
        }
#pragma unroll
        for (int v = 0; v < 8; ++v) lrow[v] += psum[v];

        // Re-read P in A-operand layout (transpose via same-wave LDS, which
        // stays in-order on DScnt).
        v16h pa;
#pragma unroll
        for (int v = 0; v < 8; ++v) {
            const int jp = apat(v, hi);
            pa[2 * v]     = lds_p[w][lo][jp];
            pa[2 * v + 1] = lds_p[w][lo][jp + 1];
        }

        // O += P * V from LDS: lane = channel, elements = keys.
#pragma unroll
        for (int t = 0; t < 16; ++t) {
            v16h vbt;
#pragma unroll
            for (int e = 0; e < 16; ++e) {
                const int kk = (e < 8) ? (8 * hi + e) : (16 + 8 * hi + (e - 8));
                vbt[e] = lds_v[kk][t * 16 + lo];
            }
            O[t] = wmma_f16(pa, vbt, O[t]);
        }
    }

    float rl[8];
#pragma unroll
    for (int v = 0; v < 8; ++v) rl[v] = 1.0f / lrow[v];
    const float g = gamma[0];

    // Epilogue: lane = channel c = t*16+lo; rows v -> consecutive n.
#pragma unroll
    for (int t = 0; t < 16; ++t) {
        const int c = t * 16 + lo;
        const size_t base = ((size_t)b * CH + c) * NPIX + n0 + 8 * hi;
        const f4* xp = (const f4*)(x + base);
        f4 x0 = xp[0], x1 = xp[1];
        f4 r0, r1;
#pragma unroll
        for (int v = 0; v < 4; ++v) r0[v] = g * (O[t][v]     * rl[v])     + x0[v];
#pragma unroll
        for (int v = 0; v < 4; ++v) r1[v] = g * (O[t][4 + v] * rl[4 + v]) + x1[v];
        f4* op = (f4*)(out + base);
        op[0] = r0;
        op[1] = r1;
    }
}

extern "C" void kernel_launch(void* const* d_in, const int* in_sizes, int n_in,
                              void* d_out, int out_size, void* d_ws, size_t ws_size,
                              hipStream_t stream) {
    (void)in_sizes; (void)n_in; (void)out_size; (void)ws_size;
    const float* x  = (const float*)d_in[0];
    const float* Wq = (const float*)d_in[1];
    const float* bq = (const float*)d_in[2];
    const float* Wk = (const float*)d_in[3];
    const float* bk = (const float*)d_in[4];
    const float* Wv = (const float*)d_in[5];
    const float* bv = (const float*)d_in[6];
    const float* gm = (const float*)d_in[7];
    float* out = (float*)d_out;

    // Workspace (f16): qT 2MB | kF 2MB | vT 16MB  (20MB total, L2-resident)
    _Float16* qT = (_Float16*)d_ws;
    _Float16* kF = qT + (size_t)BATCH * NPIX * CQK;
    _Float16* vT = kF + (size_t)BATCH * NPIX * CQK;

    const int total_waves = BATCH * NT;        // 2048 waves, 16 query rows each
    dim3 block(128);                           // 4 waves / workgroup (wave32)
    dim3 grid(total_waves / 4);                // 512 workgroups, same-batch WGs

    qkv_proj_kernel<<<grid, block, 0, stream>>>(x, Wq, bq, Wk, bk, Wv, bv,
                                                qT, kF, vT);
    flash_attn_kernel<<<grid, block, 0, stream>>>(x, qT, kF, vT, gm, out);
}
